// SuperPhysicsGNN_39754217292549
// MI455X (gfx1250) — matrix operand order repaired and checked by reading, not compiled, not run on hardware
//
#include <hip/hip_runtime.h>
#include <hip/hip_bf16.h>

// ---------------------------------------------------------------------------
// SuperPhysicsGNN for MI455X (gfx1250): 3-layer edge-featured GAT.
// Dense GEMMs: v_wmma_f32_16x16x32_bf16, wave32, 16x32 C per wave (2 tiles,
// A-fragment reuse), compile-time K (full unroll) + compile-time epilogue.
// A fragments: two contiguous b128 loads per lane (row-major A gives each lane
// one contiguous 16-element K run, k = 16*half + s, same order as B). B is pre-transposed
// to column-major bf16 so each lane's B fragment is one contiguous 32B run.
// Edge softmax/aggregation: VMEM + f32 atomics (memory-bound part).
// ---------------------------------------------------------------------------

#define N_NODES   20000
#define E_EDGES   320000
#define EF_EDGES  (E_EDGES + N_NODES)   // edges + self loops
#define IN_DIM    64

typedef __attribute__((ext_vector_type(16))) __bf16 v16bf;
typedef __attribute__((ext_vector_type(8)))  __bf16 v8bf;
typedef __attribute__((ext_vector_type(8)))  float  v8f;

// ------------------------- small device helpers ----------------------------

__device__ __forceinline__ unsigned fmap(float f) {
    unsigned u = __float_as_uint(f);
    return (u & 0x80000000u) ? ~u : (u | 0x80000000u);
}
__device__ __forceinline__ float funmap(unsigned m) {
    unsigned u = (m & 0x80000000u) ? (m & 0x7FFFFFFFu) : ~m;
    return __uint_as_float(u);
}
__device__ __forceinline__ void edge_nodes(const int* ei, int e, int& s, int& d) {
    if (e < E_EDGES) { s = ei[e]; d = ei[E_EDGES + e]; }
    else             { s = d = e - E_EDGES; }
}
__device__ __forceinline__ float wave_sum(float v) {
    #pragma unroll
    for (int m = 16; m; m >>= 1) v += __shfl_xor(v, m, 32);
    return v;
}

// ------------------------- self-loop attr (fill_value='mean') --------------

__global__ void k_loop_accum(const int* ei, const float* ea, float* cnt, float* lsum) {
    int e = blockIdx.x * blockDim.x + threadIdx.x;
    if (e >= E_EDGES) return;
    int d = ei[E_EDGES + e];
    atomicAdd(&cnt[d], 1.0f);
    atomicAdd(&lsum[d * 3 + 0], ea[e * 3 + 0]);
    atomicAdd(&lsum[d * 3 + 1], ea[e * 3 + 1]);
    atomicAdd(&lsum[d * 3 + 2], ea[e * 3 + 2]);
}

__global__ void k_loop_div(float* lsum, const float* cnt) {
    int t = blockIdx.x * blockDim.x + threadIdx.x;
    if (t >= N_NODES * 3) return;
    lsum[t] /= fmaxf(cnt[t / 3], 1.0f);
}

// ------------------------- input LayerNorm -> bf16 -------------------------

__global__ void k_ln_in(const float* x, const float* g, const float* b, __bf16* out) {
    int wid  = blockIdx.x * (blockDim.x >> 5) + (threadIdx.x >> 5);
    int lane = threadIdx.x & 31;
    if (wid >= N_NODES) return;
    const float* xp = x + (size_t)wid * IN_DIM;
    float v0 = xp[lane], v1 = xp[lane + 32];
    float mean = wave_sum(v0 + v1) * (1.0f / IN_DIM);
    float d0 = v0 - mean, d1 = v1 - mean;
    float var  = wave_sum(d0 * d0 + d1 * d1) * (1.0f / IN_DIM);
    float rinv = rsqrtf(var + 1e-5f);
    out[(size_t)wid * IN_DIM + lane]      = (__bf16)(d0 * rinv * g[lane] + b[lane]);
    out[(size_t)wid * IN_DIM + lane + 32] = (__bf16)(d1 * rinv * g[lane + 32] + b[lane + 32]);
}

// ------------------------- converts ----------------------------------------

__global__ void k_f32_to_bf16(const float* src, __bf16* dst, int n) {
    int i = blockIdx.x * blockDim.x + threadIdx.x;
    if (i < n) dst[i] = (__bf16)src[i];
}

// weight [K x Nout] f32 row-major -> bf16 column-major Bt[Nout x K]
__global__ void k_wT_bf16(const float* W, __bf16* Bt, int K, int Nout) {
    int t = blockIdx.x * blockDim.x + threadIdx.x;
    if (t >= K * Nout) return;
    int k = t / Nout, n = t % Nout;
    Bt[(size_t)n * K + k] = (__bf16)W[t];
}

// ------------------------- WMMA bf16 GEMM ----------------------------------
// C[M x Nout] = A[M x K] @ B[K x Nout]; A row-major bf16, B given transposed
// (Bt[Nout x K] bf16). One wave computes a 16x32 strip of C (two 16x16 tiles
// sharing the A fragment). KC compile-time (full unroll), MODE compile-time:
// bit0 +bias[col], bit1 relu, bit2 accumulate into C.

__device__ __forceinline__ v8f wmma_bf16(v16bf a, v16bf b, v8f c) {
    return __builtin_amdgcn_wmma_f32_16x16x32_bf16(false, a, false, b,
                                                   (short)0, c, false, false);
}

template <int KC, int MODE>
__global__ __launch_bounds__(256)
void k_gemm(const __bf16* __restrict__ A, const __bf16* __restrict__ Bt,
            const float* __restrict__ bias, float* __restrict__ C,
            int M, int Nout) {
    int wave = blockIdx.x * (blockDim.x >> 5) + (threadIdx.x >> 5);
    int lane = threadIdx.x & 31;
    int pairsN = Nout >> 5;                     // 16x32 strips along N
    int nstrip = (M >> 4) * pairsN;
    if (wave >= nstrip) return;                 // wave-uniform: EXEC all-1s
    int tm = wave / pairsN, tp = wave % pairsN;
    int half = lane >> 4, l15 = lane & 15;

    const __bf16* ap  = A  + (size_t)(tm * 16 + l15) * KC + (half << 4);
    const __bf16* b0p = Bt + (size_t)(tp * 32 + l15) * KC + (half << 4);
    const __bf16* b1p = b0p + (size_t)16 * KC;

    v8f acc0 = {}, acc1 = {};
    #pragma unroll
    for (int kk = 0; kk < KC; kk += 32) {
        // A 16x32 frag: lane holds a contiguous 16-elem run of K (k = 16*half + s)
        v8bf alo = *(const v8bf*)(ap + kk);
        v8bf ahi = *(const v8bf*)(ap + kk + 8);
        v16bf a = __builtin_shufflevector(alo, ahi,
                    0,1,2,3,4,5,6,7,8,9,10,11,12,13,14,15);
        // B 32x16 frag: contiguous 16-elem run of K in column-major Bt
        v16bf b0 = *(const v16bf*)(b0p + kk);
        v16bf b1 = *(const v16bf*)(b1p + kk);
        acc0 = wmma_bf16(a, b0, acc0);
        acc1 = wmma_bf16(a, b1, acc1);
    }

    int col0 = tp * 32 + l15;
    int col1 = col0 + 16;
    float bia0 = 0.0f, bia1 = 0.0f;
    if (MODE & 1) { bia0 = bias[col0]; bia1 = bias[col1]; }
    #pragma unroll
    for (int r = 0; r < 8; ++r) {
        int row = tm * 16 + r + (half << 3);    // C: VGPR r -> M = r + 8*half
        float v0 = acc0[r] + bia0;
        float v1 = acc1[r] + bia1;
        if (MODE & 2) { v0 = fmaxf(v0, 0.0f); v1 = fmaxf(v1, 0.0f); }
        float* cp = &C[(size_t)row * Nout + col0];
        if (MODE & 4) { cp[0] += v0; cp[16] += v1; }
        else          { cp[0]  = v0; cp[16]  = v1; }
    }
}

// ------------------------- attention logits --------------------------------

__global__ void k_eM(const float* We, const float* ae, float* eM, int H, int O, int F) {
    int t = threadIdx.x;                        // 3*H <= 12 threads
    if (t >= 3 * H) return;
    int d = t / H, h = t % H;
    float s = 0.0f;
    for (int o = 0; o < O; ++o) s += We[d * F + h * O + o] * ae[h * O + o];
    eM[d * 4 + h] = s;                          // eM stride 4 (max H)
}

__global__ void k_node_logits(const float* hp, const float* as, const float* ad,
                              float* al_s, float* al_d, int H, int O) {
    int t = blockIdx.x * blockDim.x + threadIdx.x;
    if (t >= N_NODES * H) return;
    int n = t / H, h = t % H;
    const float* hpp = hp + (size_t)(n * H + h) * O;
    float ss = 0.0f, sd = 0.0f;
    for (int o = 0; o < O; ++o) {
        float v = hpp[o];
        ss += v * as[h * O + o];
        sd += v * ad[h * O + o];
    }
    al_s[t] = ss; al_d[t] = sd;
}

// ------------------------- edge softmax ------------------------------------

__global__ void k_edge_alpha(const int* ei, const float* ea_f, const float* eM,
                             const float* al_s, const float* al_d,
                             float* alpha, unsigned* mbuf, int H) {
    int t = blockIdx.x * blockDim.x + threadIdx.x;
    if (t >= EF_EDGES * H) return;
    int e = t / H, h = t % H;
    int s, d; edge_nodes(ei, e, s, d);
    float ale = ea_f[e * 3 + 0] * eM[0 * 4 + h]
              + ea_f[e * 3 + 1] * eM[1 * 4 + h]
              + ea_f[e * 3 + 2] * eM[2 * 4 + h];
    float a = al_s[s * H + h] + al_d[d * H + h] + ale;
    a = (a > 0.0f) ? a : 0.2f * a;              // leaky_relu(0.2)
    alpha[t] = a;
    atomicMax(&mbuf[d * H + h], fmap(a));       // scatter-max (ordered-int)
}

__global__ void k_edge_exp(const int* ei, float* alpha, const unsigned* mbuf,
                           float* den, int H) {
    int t = blockIdx.x * blockDim.x + threadIdx.x;
    if (t >= EF_EDGES * H) return;
    int e = t / H, h = t % H;
    int s, d; edge_nodes(ei, e, s, d);
    float ex = __expf(alpha[t] - funmap(mbuf[d * H + h]));
    alpha[t] = ex;
    atomicAdd(&den[d * H + h], ex);
}

__global__ void k_edge_w(const int* ei, float* alpha, const float* den, int H) {
    int t = blockIdx.x * blockDim.x + threadIdx.x;
    if (t >= EF_EDGES * H) return;
    int e = t / H, h = t % H;
    int s, d; edge_nodes(ei, e, s, d);
    alpha[t] = alpha[t] / (den[d * H + h] + 1e-16f);
}

// ------------------------- weighted scatter aggregation --------------------
// One thread handles 4 consecutive features (never straddles a head: O%4==0).

__global__ void k_edge_agg(const int* ei, const float* alpha, const float* hp,
                           float* agg, int H, int O) {
    int F = H * O, Fq = F >> 2;
    size_t total = (size_t)EF_EDGES * Fq;
    size_t t = (size_t)blockIdx.x * blockDim.x + threadIdx.x;
    if (t >= total) return;
    int e = (int)(t / Fq), f = ((int)(t % Fq)) << 2;
    int h = f / O;
    int s, d; edge_nodes(ei, e, s, d);
    float w = alpha[e * H + h];
    const float4 v = *(const float4*)(hp + (size_t)s * F + f);
    float* apt = agg + (size_t)d * F + f;
    atomicAdd(apt + 0, w * v.x);
    atomicAdd(apt + 1, w * v.y);
    atomicAdd(apt + 2, w * v.z);
    atomicAdd(apt + 3, w * v.w);
}

// ------------------------- bias + LayerNorm + ELU -> bf16 ------------------

__global__ void k_ln_elu(const float* agg, const float* bias, const float* g,
                         const float* b, __bf16* out, int F) {
    int wid  = blockIdx.x * (blockDim.x >> 5) + (threadIdx.x >> 5);
    int lane = threadIdx.x & 31;
    if (wid >= N_NODES) return;
    int per = F >> 5;                           // F in {256,128,64} -> 8,4,2
    float vals[8];
    float s = 0.0f;
    for (int j = 0; j < per; ++j) {
        int f = lane + 32 * j;
        vals[j] = agg[(size_t)wid * F + f] + bias[f];
        s += vals[j];
    }
    float mean = wave_sum(s) / (float)F;
    float vs = 0.0f;
    for (int j = 0; j < per; ++j) { float dd = vals[j] - mean; vs += dd * dd; }
    float rinv = rsqrtf(wave_sum(vs) / (float)F + 1e-5f);
    for (int j = 0; j < per; ++j) {
        int f = lane + 32 * j;
        float y = (vals[j] - mean) * rinv * g[f] + b[f];
        y = (y > 0.0f) ? y : (__expf(y) - 1.0f);    // ELU(alpha=1)
        out[(size_t)wid * F + f] = (__bf16)y;
    }
}

// ------------------------- final row normalize -----------------------------

__global__ void k_norm(float* out) {
    int wid  = blockIdx.x * (blockDim.x >> 5) + (threadIdx.x >> 5);
    int lane = threadIdx.x & 31;
    if (wid >= N_NODES) return;
    float v = out[(size_t)wid * 32 + lane];
    float nrm = sqrtf(wave_sum(v * v));
    out[(size_t)wid * 32 + lane] = v / fmaxf(nrm, 1e-12f);
}

// ---------------------------------------------------------------------------

static inline unsigned cdivu(size_t a, size_t b) { return (unsigned)((a + b - 1) / b); }

static void gemm_launch(int K, int mode,
                        const __bf16* A, const __bf16* Bt, const float* bias,
                        float* C, int M, int Nout, hipStream_t s) {
    unsigned blocks = cdivu((size_t)(M / 16) * (Nout / 32), 8);
    #define GEMM_CASE(KK, MM) \
        if (K == KK && mode == MM) { \
            k_gemm<KK, MM><<<blocks, 256, 0, s>>>(A, Bt, bias, C, M, Nout); return; }
    GEMM_CASE(64, 3)
    GEMM_CASE(64, 0)
    GEMM_CASE(256, 0)
    GEMM_CASE(128, 0)
    GEMM_CASE(32, 1)
    GEMM_CASE(64, 5)
    #undef GEMM_CASE
}

extern "C" void kernel_launch(void* const* d_in, const int* in_sizes, int n_in,
                              void* d_out, int out_size, void* d_ws, size_t ws_size,
                              hipStream_t stream) {
    (void)in_sizes; (void)n_in; (void)out_size; (void)ws_size;

    const float* x        = (const float*)d_in[0];
    const float* eattr    = (const float*)d_in[1];
    const float* in_ln_g  = (const float*)d_in[2];
    const float* in_ln_b  = (const float*)d_in[3];
    const float* proj_W   = (const float*)d_in[4];
    const float* proj_b   = (const float*)d_in[5];
    const float* Wp[3]    = {(const float*)d_in[6],  (const float*)d_in[14], (const float*)d_in[22]};
    const float* Wep[3]   = {(const float*)d_in[7],  (const float*)d_in[15], (const float*)d_in[23]};
    const float* asp[3]   = {(const float*)d_in[8],  (const float*)d_in[16], (const float*)d_in[24]};
    const float* adp[3]   = {(const float*)d_in[9],  (const float*)d_in[17], (const float*)d_in[25]};
    const float* aep[3]   = {(const float*)d_in[10], (const float*)d_in[18], (const float*)d_in[26]};
    const float* biasp[3] = {(const float*)d_in[11], (const float*)d_in[19], (const float*)d_in[27]};
    const float* lngp[3]  = {(const float*)d_in[12], (const float*)d_in[20], (const float*)d_in[28]};
    const float* lnbp[3]  = {(const float*)d_in[13], (const float*)d_in[21], (const float*)d_in[29]};
    const float* out_W1   = (const float*)d_in[30];
    const float* out_b1   = (const float*)d_in[31];
    const float* out_W2   = (const float*)d_in[32];
    const float* out_b2   = (const float*)d_in[33];
    const float* skip_W   = (const float*)d_in[34];
    const float* skip_b   = (const float*)d_in[35];
    const int*   ei       = (const int*)d_in[36];
    float*       outp     = (float*)d_out;

    // ---- workspace bump allocator ----
    char* wsp = (char*)d_ws;
    auto alloc = [&](size_t bytes) -> void* {
        void* p = wsp; wsp += (bytes + 255) & ~(size_t)255; return p;
    };
    float*    initf = (float*)   alloc((size_t)N_NODES * 64 * 4);
    __bf16*   ibf   = (__bf16*)  alloc((size_t)N_NODES * 64 * 2);
    __bf16*   xlnbf = (__bf16*)  alloc((size_t)N_NODES * 64 * 2);
    __bf16*   hbf   = (__bf16*)  alloc((size_t)N_NODES * 256 * 2);
    float*    hp    = (float*)   alloc((size_t)N_NODES * 256 * 4);
    float*    agg   = (float*)   alloc((size_t)N_NODES * 256 * 4);
    float*    al_s  = (float*)   alloc((size_t)N_NODES * 4 * 4);
    float*    al_d  = (float*)   alloc((size_t)N_NODES * 4 * 4);
    unsigned* mbuf  = (unsigned*)alloc((size_t)N_NODES * 4 * 4);
    float*    den   = (float*)   alloc((size_t)N_NODES * 4 * 4);
    float*    cnt   = (float*)   alloc((size_t)N_NODES * 4);
    float*    ea_f  = (float*)   alloc((size_t)EF_EDGES * 3 * 4);
    float*    alpha = (float*)   alloc((size_t)EF_EDGES * 4 * 4);
    __bf16*   wbf   = (__bf16*)  alloc((size_t)256 * 256 * 2);
    float*    eM    = (float*)   alloc(64);
    float*    t1f   = (float*)   alloc((size_t)N_NODES * 32 * 4);
    __bf16*   t1bf  = (__bf16*)  alloc((size_t)N_NODES * 32 * 2);

    const int BLK = 256;

    // ---- self-loop edge_attr = mean of incoming edge_attr ----
    hipMemsetAsync(cnt, 0, (size_t)N_NODES * 4, stream);
    hipMemsetAsync(ea_f + (size_t)3 * E_EDGES, 0, (size_t)N_NODES * 3 * 4, stream);
    hipMemcpyAsync(ea_f, eattr, (size_t)E_EDGES * 3 * 4, hipMemcpyDeviceToDevice, stream);
    k_loop_accum<<<cdivu(E_EDGES, BLK), BLK, 0, stream>>>(ei, eattr, cnt, ea_f + (size_t)3 * E_EDGES);
    k_loop_div<<<cdivu((size_t)N_NODES * 3, BLK), BLK, 0, stream>>>(ea_f + (size_t)3 * E_EDGES, cnt);

    // ---- input LN + projection (WMMA) ----
    k_ln_in<<<cdivu(N_NODES, 8), BLK, 0, stream>>>(x, in_ln_g, in_ln_b, xlnbf);
    k_wT_bf16<<<cdivu(64 * 64, BLK), BLK, 0, stream>>>(proj_W, wbf, 64, 64);
    gemm_launch(64, 3, xlnbf, wbf, proj_b, initf, N_NODES, 64, stream);
    k_f32_to_bf16<<<cdivu((size_t)N_NODES * 64, BLK), BLK, 0, stream>>>(initf, ibf, N_NODES * 64);

    // ---- 3 GAT layers ----
    const int Hh[3] = {4, 2, 2}, Oo[3] = {64, 64, 32}, Kin[3] = {64, 256, 128};
    for (int i = 0; i < 3; ++i) {
        int H = Hh[i], O = Oo[i], F = H * O, K = Kin[i];
        const __bf16* Abf = (i == 0) ? ibf : hbf;

        k_wT_bf16<<<cdivu((size_t)K * F, BLK), BLK, 0, stream>>>(Wp[i], wbf, K, F);
        gemm_launch(K, 0, Abf, wbf, nullptr, hp, N_NODES, F, stream);

        k_eM<<<1, 32, 0, stream>>>(Wep[i], aep[i], eM, H, O, F);
        k_node_logits<<<cdivu((size_t)N_NODES * H, BLK), BLK, 0, stream>>>(
            hp, asp[i], adp[i], al_s, al_d, H, O);

        hipMemsetAsync(mbuf, 0, (size_t)N_NODES * H * 4, stream);   // below fmap(-max)
        hipMemsetAsync(den, 0, (size_t)N_NODES * H * 4, stream);
        k_edge_alpha<<<cdivu((size_t)EF_EDGES * H, BLK), BLK, 0, stream>>>(
            ei, ea_f, eM, al_s, al_d, alpha, mbuf, H);
        k_edge_exp<<<cdivu((size_t)EF_EDGES * H, BLK), BLK, 0, stream>>>(ei, alpha, mbuf, den, H);
        k_edge_w<<<cdivu((size_t)EF_EDGES * H, BLK), BLK, 0, stream>>>(ei, alpha, den, H);

        hipMemsetAsync(agg, 0, (size_t)N_NODES * F * 4, stream);
        k_edge_agg<<<cdivu((size_t)EF_EDGES * (F >> 2), BLK), BLK, 0, stream>>>(
            ei, alpha, hp, agg, H, O);

        k_ln_elu<<<cdivu(N_NODES, 8), BLK, 0, stream>>>(agg, biasp[i], lngp[i], lnbp[i], hbf, F);
    }

    // ---- output MLP + skip + normalize (WMMA) ----
    k_wT_bf16<<<cdivu(64 * 32, BLK), BLK, 0, stream>>>(out_W1, wbf, 64, 32);
    gemm_launch(64, 3, hbf, wbf, out_b1, t1f, N_NODES, 32, stream);
    k_f32_to_bf16<<<cdivu((size_t)N_NODES * 32, BLK), BLK, 0, stream>>>(t1f, t1bf, N_NODES * 32);

    k_wT_bf16<<<cdivu(32 * 32, BLK), BLK, 0, stream>>>(out_W2, wbf, 32, 32);
    gemm_launch(32, 1, t1bf, wbf, out_b2, outp, N_NODES, 32, stream);

    k_wT_bf16<<<cdivu(64 * 32, BLK), BLK, 0, stream>>>(skip_W, wbf, 64, 32);
    gemm_launch(64, 5, ibf, wbf, skip_b, outp, N_NODES, 32, stream);

    k_norm<<<cdivu(N_NODES, 8), BLK, 0, stream>>>(outp);
}
